// FFTMambaRegressor_55825984913781
// MI455X (gfx1250) — matrix operand order repaired and checked by reading, not compiled, not run on hardware
//
#include <hip/hip_runtime.h>
#include <hip/hip_bf16.h>

// ---------------------------------------------------------------------------
// FFT-Mamba regressor for gfx1250 (MI455X).
// GEMMs run on v_wmma_f32_16x16x32_bf16 (bf16 A/B, f32 accum); A tiles are
// staged LDS-side by the Tensor Data Mover (tensor_load_to_lds + TENSORcnt).
// Everything else (LN, FFT, conv, selective scan, head) stays f32.
// ---------------------------------------------------------------------------

typedef __attribute__((ext_vector_type(16))) __bf16 v16bf;
typedef __attribute__((ext_vector_type(8)))  float  v8f;
typedef __attribute__((ext_vector_type(4)))  unsigned int v4u;
typedef __attribute__((ext_vector_type(4)))  int v4i;
typedef __attribute__((ext_vector_type(8)))  int v8i;

#define FM_HAS_TDM __has_builtin(__builtin_amdgcn_tensor_load_to_lds)

namespace {
constexpr int kBatch = 64;
constexpr int kCtx   = 4096;
constexpr int kNP    = 64;       // patches
constexpr int kPatch = 64;
constexpr int kDM    = 1024;
constexpr int kDI    = 2048;
constexpr int kDS    = 16;
constexpr int kDTR   = 64;
constexpr int kL     = 128;      // 2 * n_patches (x tokens ++ f tokens)
constexpr int kTok   = kBatch * kL;   // 8192 rows for all big GEMMs
constexpr float kPi  = 3.14159265358979323846f;
}

__device__ __forceinline__ __bf16 f2bf(float f) {
  union { float f; unsigned u; } v; v.f = f;
  unsigned r = v.u + 0x7FFFu + ((v.u >> 16) & 1u);   // round-to-nearest-even
  unsigned short h = (unsigned short)(r >> 16);
  return __builtin_bit_cast(__bf16, h);
}
__device__ __forceinline__ float sigmoidf_(float x) { return 1.f / (1.f + __expf(-x)); }

// ---------------------------------------------------------------------------
// f32 -> bf16 conversion (weights + activations)
// ---------------------------------------------------------------------------
__global__ __launch_bounds__(256) void fm_to_bf16(const float* __restrict__ in,
                                                  __bf16* __restrict__ out, size_t n) {
  size_t i = (size_t)blockIdx.x * 256 + threadIdx.x;
  if (i < n) out[i] = f2bf(in[i]);
}

// ---------------------------------------------------------------------------
// Generic bf16 WMMA GEMM: C(MxN) = A(MxK,row) * B(KxN,row)  [+bias] [softplus]
// Block = 256 threads (8 waves). Tile: 128x64, K-step 32.
// Wave (wm,wn): 64x16 strip = 4 v_wmma_f32_16x16x32_bf16 sharing one B frag.
// A tile staged by TDM (wave 0, uniform 2D descriptor, TENSORcnt-drained);
// B tile staged transposed so fragments are contiguous 32B ds_load_b128 pairs.
// OP: 0 = none, 1 = +bias, 2 = +bias then softplus.  GUARD: N not mult of 64.
// Requires: M % 128 == 0, K % 32 == 0.
// ---------------------------------------------------------------------------
#define GBM 128
#define GBN 64
#define GBK 32

template <int OP, bool GUARD>
__global__ __launch_bounds__(256) void fm_gemm_bf16(
    const __bf16* __restrict__ A, const __bf16* __restrict__ B, float* __restrict__ C,
    const float* __restrict__ bias, int M, int N, int K) {
  __shared__ __align__(16) __bf16 As[GBM * GBK];     // [row][k]   row-major
  __shared__ __align__(16) __bf16 Bs[GBN * GBK];     // [col][k]   transposed!
  const int tid  = threadIdx.x;
  const int lane = tid & 31;
  const int wid  = tid >> 5;
  const int hf   = lane >> 4;      // wave32 half-select per ISA layout tables
  const int ln16 = lane & 15;
  const int bm = blockIdx.x * GBM;
  const int bn = blockIdx.y * GBN;
  const int wn = (wid >> 1) * 16;  // 4 waves across N
  const int wm = (wid & 1) * 64;   // 2 waves across M, 4 tiles each

  v8f acc[4];
#pragma unroll
  for (int t = 0; t < 4; ++t)
#pragma unroll
    for (int v = 0; v < 8; ++v) acc[t][v] = 0.f;

  for (int k0 = 0; k0 < K; k0 += GBK) {
    // ---- stage A tile (128 rows x 32 k, bf16) -----------------------------
#if FM_HAS_TDM
    if (tid < 32) {
      // 2D TDM descriptor: data_size=2B, tile 32x128, row stride = K elements.
      unsigned long long ga =
          (unsigned long long)(const void*)(A + ((size_t)bm * K + k0));
      unsigned lds = (unsigned)(unsigned long long)(uintptr_t)(&As[0]);
      v4u g0;
      g0[0] = 1u;                                   // count=1 (valid user D#)
      g0[1] = lds;                                  // lds_addr
      g0[2] = (unsigned)(ga & 0xFFFFFFFFu);         // global_addr[31:0]
      g0[3] = (unsigned)((ga >> 32) & 0x1FFFFFFu)   // global_addr[56:32]
              | (2u << 30);                         // type = 2 ("image")
      v8i g1;
      g1[0] = 0x10000;                              // data_size = 2 bytes
      g1[1] = (int)((K & 0xFFFF) << 16);            // tensor_dim0 lo
      g1[2] = (int)(((unsigned)K >> 16) & 0xFFFFu) |
              (int)((M & 0xFFFF) << 16);            // dim0 hi | tensor_dim1 lo
      g1[3] = (int)(((unsigned)M >> 16) & 0xFFFFu) |
              (GBK << 16);                          // dim1 hi | tile_dim0 = 32
      g1[4] = GBM;                                  // tile_dim1 = 128
      g1[5] = K;                                    // tensor_dim0_stride lo
      g1[6] = 0;                                    // stride hi | dim1_stride lo
      g1[7] = 0;
      v4i gz = {0, 0, 0, 0};
#if __clang_major__ >= 23
      __builtin_amdgcn_tensor_load_to_lds(g0, g1, gz, gz, (v8i)(0), 0);
#else
      __builtin_amdgcn_tensor_load_to_lds(g0, g1, gz, gz, 0);
#endif
    }
#else
#pragma unroll
    for (int i = 0; i < 2; ++i) {
      int e = (tid + i * 256) * 8;
      int r = e >> 5, c = e & 31;
      *(float4*)(&As[e]) = *(const float4*)(&A[(size_t)(bm + r) * K + k0 + c]);
    }
#endif
    // ---- stage B tile (32x64 bf16) transposed into Bs[col][k] -------------
    {
      int e = tid * 8;
      int r = e >> 6, cc = e & 63;                  // r = k row, cc = col base
      int gc = bn + cc;
      if (!GUARD || gc + 8 <= N) {
        float4 tmp = *(const float4*)(&B[(size_t)(k0 + r) * N + gc]);
        const __bf16* tb = (const __bf16*)&tmp;
#pragma unroll
        for (int j = 0; j < 8; ++j) Bs[(cc + j) * GBK + r] = tb[j];
      } else {
#pragma unroll
        for (int j = 0; j < 8; ++j)
          Bs[(cc + j) * GBK + r] = (gc + j < N) ? B[(size_t)(k0 + r) * N + gc + j]
                                                : f2bf(0.f);
      }
    }
    if (k0 + GBK < K)
      __builtin_prefetch(&B[(size_t)(k0 + GBK + (tid & 31)) * N + bn], 0, 1);
#if FM_HAS_TDM
    if (tid < 32) __builtin_amdgcn_s_wait_tensorcnt(0);   // drain wave-0 TDM
#endif
    __syncthreads();

    // ---- B fragment (32x16): lanes 0-15 K=0..15, lanes 16-31 K=16..31.
    // Transposed Bs makes this 32 contiguous bytes per lane.
    v16bf bf;
    {
      const __bf16* bp = &Bs[(wn + ln16) * GBK + hf * 16];
#pragma unroll
      for (int j = 0; j < 16; ++j) bf[j] = bp[j];
    }

#pragma unroll
    for (int t = 0; t < 4; ++t) {
      // ---- A fragment (16x32), contiguous K order (matches B fragment):
      // lane L(0-15): row M=L, K0..15;
      // lane L+16: same row, K16..31.
      const __bf16* ap = &As[(wm + t * 16) * GBK];
      v16bf af;
#pragma unroll
      for (int v = 0; v < 8; ++v) {
        int kk = 2 * v + hf * 16;
        af[2 * v]     = ap[ln16 * GBK + kk];
        af[2 * v + 1] = ap[ln16 * GBK + kk + 1];
      }
      acc[t] = __builtin_amdgcn_wmma_f32_16x16x32_bf16(
          false, af, false, bf, (short)0, acc[t], false, false);
    }
    __syncthreads();
  }

  // ---- epilogue: C/D layout = VGPR v: lanes 0-15 M=v, lanes 16-31 M=v+8 ----
  const int col = bn + wn + ln16;
  if (!GUARD || col < N) {
#pragma unroll
    for (int t = 0; t < 4; ++t) {
#pragma unroll
      for (int v = 0; v < 8; ++v) {
        int row = bm + wm + t * 16 + v + hf * 8;
        float val = acc[t][v];
        if (OP >= 1) val += bias[col];
        if (OP == 2) val = (val > 20.f) ? val : log1pf(__expf(val));
        C[(size_t)row * N + col] = val;
      }
    }
  }
}

// ---------------------------------------------------------------------------
// LayerNorm over small rows (W = 64 or 128), output bf16 (GEMM A operand).
// One block of W threads per row.
// ---------------------------------------------------------------------------
__global__ __launch_bounds__(128) void fm_ln_small(
    const float* __restrict__ x, const float* __restrict__ g, const float* __restrict__ b,
    __bf16* __restrict__ out, int W) {
  __shared__ float sb[128];
  const int row = blockIdx.x;
  const int t = threadIdx.x;
  float v = x[(size_t)row * W + t];
  sb[t] = v; __syncthreads();
  for (int o = W >> 1; o > 0; o >>= 1) { if (t < o) sb[t] += sb[t + o]; __syncthreads(); }
  float mu = sb[0] / (float)W;
  __syncthreads();
  float d = v - mu;
  sb[t] = d * d; __syncthreads();
  for (int o = W >> 1; o > 0; o >>= 1) { if (t < o) sb[t] += sb[t + o]; __syncthreads(); }
  float rs = rsqrtf(sb[0] / (float)W + 1e-5f);
  out[(size_t)row * W + t] = f2bf(d * rs * g[t] + b[t]);
}

// ---------------------------------------------------------------------------
// 4096-point radix-2 iterative FFT per batch row, fully in LDS (32 KB).
// Writes interleaved (re,im) -> (B, 4096, 2) == (B, 64, 128) patch rows.
// ---------------------------------------------------------------------------
__global__ __launch_bounds__(256) void fm_fft4096(const float* __restrict__ x,
                                                  float* __restrict__ fr) {
  __shared__ float re[4096];
  __shared__ float im[4096];
  const int b = blockIdx.x, tid = threadIdx.x;
  for (int i = tid; i < 4096; i += 256) {
    int r = (int)(__brev((unsigned)i) >> 20);   // 12-bit bit-reversal
    re[r] = x[(size_t)b * 4096 + i];
    im[r] = 0.f;
  }
  __syncthreads();
  for (int s = 1; s <= 12; ++s) {
    const int m = 1 << s, mh = m >> 1;
    for (int j = tid; j < 2048; j += 256) {
      int grp = j / mh, pos = j - grp * mh;
      int i0 = grp * m + pos, i1 = i0 + mh;
      float ang = -2.f * kPi * (float)pos / (float)m;
      float sn, c; __sincosf(ang, &sn, &c);
      float tr = re[i1] * c - im[i1] * sn;
      float ti = re[i1] * sn + im[i1] * c;
      re[i1] = re[i0] - tr; im[i1] = im[i0] - ti;
      re[i0] += tr;         im[i0] += ti;
    }
    __syncthreads();
  }
  for (int i = tid; i < 4096; i += 256) {
    fr[((size_t)b * 4096 + i) * 2 + 0] = re[i];
    fr[((size_t)b * 4096 + i) * 2 + 1] = im[i];
  }
}

// ---------------------------------------------------------------------------
// Second layernorm (over d_model=1024) + sincos positional embedding.
// Writes both f32 and bf16 copies of the concatenated token matrix.
// part: 0 = patch tokens (L slot n), 1 = freq tokens (L slot 64+n).
// ---------------------------------------------------------------------------
__global__ __launch_bounds__(256) void fm_ln_pe(
    const float* __restrict__ in, const float* __restrict__ g, const float* __restrict__ bta,
    float* __restrict__ xcat, __bf16* __restrict__ xcatb, int part) {
  __shared__ float sb[256];
  const int row = blockIdx.x;          // 0..4095  (b*64 + n)
  const int b = row >> 6, n = row & 63;
  const float* x = in + (size_t)row * kDM;
  float vals[4];
  float s = 0.f;
#pragma unroll
  for (int k = 0; k < 4; ++k) { vals[k] = x[threadIdx.x + k * 256]; s += vals[k]; }
  sb[threadIdx.x] = s; __syncthreads();
  for (int o = 128; o > 0; o >>= 1) { if (threadIdx.x < o) sb[threadIdx.x] += sb[threadIdx.x + o]; __syncthreads(); }
  float mu = sb[0] * (1.f / 1024.f);
  __syncthreads();
  s = 0.f;
#pragma unroll
  for (int k = 0; k < 4; ++k) { float d = vals[k] - mu; s += d * d; }
  sb[threadIdx.x] = s; __syncthreads();
  for (int o = 128; o > 0; o >>= 1) { if (threadIdx.x < o) sb[threadIdx.x] += sb[threadIdx.x + o]; __syncthreads(); }
  float rs = rsqrtf(sb[0] * (1.f / 1024.f) + 1e-5f);
  const size_t obase = ((size_t)(b * kL + part * kNP + n)) * kDM;
#pragma unroll
  for (int k = 0; k < 4; ++k) {
    int d = threadIdx.x + k * 256;
    float y = (vals[k] - mu) * rs * g[d] + bta[d];
    int i = (d < 512) ? d : d - 512;
    float omega = __expf(-9.210340372f * (float)i / 511.f);  // 1/10000^(i/511)
    float ang = (float)n * omega;
    y += (d < 512) ? __sinf(ang) : __cosf(ang);
    xcat[obase + d]  = y;
    xcatb[obase + d] = f2bf(y);
  }
}

// ---------------------------------------------------------------------------
// Depthwise causal conv1d (K=4) over L + SiLU. xs = xz[..., :2048].
// ---------------------------------------------------------------------------
__global__ __launch_bounds__(256) void fm_conv_silu(
    const float* __restrict__ xz, const float* __restrict__ cw, const float* __restrict__ cb,
    float* __restrict__ xc, __bf16* __restrict__ xcb) {
  size_t i = (size_t)blockIdx.x * 256 + threadIdx.x;   // over 64*128*2048
  int c = (int)(i & 2047);
  int t = (int)((i >> 11) & 127);
  int b = (int)(i >> 18);
  const float* base = xz + ((size_t)b * kL) * (2 * kDI) + c;
  float acc = cb[c];
#pragma unroll
  for (int k = 0; k < 4; ++k) {
    int tt = t - 3 + k;
    if (tt >= 0) acc += cw[c * 4 + k] * base[(size_t)tt * (2 * kDI)];
  }
  float sv = acc * sigmoidf_(acc);
  xc[i] = sv; xcb[i] = f2bf(sv);
}

// ---------------------------------------------------------------------------
// Slice dt-rank columns of x_dbl (8192 x 96) into bf16 (8192 x 64) for dt_proj.
// ---------------------------------------------------------------------------
__global__ __launch_bounds__(256) void fm_split_dt(const float* __restrict__ xdbl,
                                                   __bf16* __restrict__ dtb) {
  size_t i = (size_t)blockIdx.x * 256 + threadIdx.x;   // over 8192*64
  int col = (int)(i & 63);
  size_t row = i >> 6;
  dtb[i] = f2bf(xdbl[row * 96 + col]);
}

// ---------------------------------------------------------------------------
// Selective scan. One thread per (batch, channel); h[16] in VGPRs;
// B_t/C_t (16 KB) staged in LDS once per block. Fused +xc*D and *silu(z).
// grid = (8, 64): blockIdx.y = batch, blockIdx.x = channel chunk of 256.
// ---------------------------------------------------------------------------
__global__ __launch_bounds__(256) void fm_scan(
    const float* __restrict__ dt, const float* __restrict__ xc,
    const float* __restrict__ xdbl, const float* __restrict__ z,
    const float* __restrict__ A_log, const float* __restrict__ Dp,
    float* __restrict__ y, __bf16* __restrict__ yb) {
  __shared__ float Bs[kL * kDS];
  __shared__ float Cs[kL * kDS];
  const int b = blockIdx.y;
  const int c = blockIdx.x * 256 + threadIdx.x;
  for (int i = threadIdx.x; i < kL * kDS; i += 256) {
    int t = i >> 4, j = i & 15;
    const float* rowp = xdbl + ((size_t)(b * kL + t)) * 96;
    Bs[i] = rowp[64 + j];
    Cs[i] = rowp[80 + j];
  }
  __syncthreads();
  float A[kDS], h[kDS];
#pragma unroll
  for (int j = 0; j < kDS; ++j) { A[j] = -__expf(A_log[(size_t)c * kDS + j]); h[j] = 0.f; }
  const float Dc = Dp[c];
  for (int t = 0; t < kL; ++t) {
    size_t idx = ((size_t)(b * kL + t)) * kDI + c;
    float dtv = dt[idx];
    float xv  = xc[idx];
    float dtx = dtv * xv;
    float yv = 0.f;
#pragma unroll
    for (int j = 0; j < kDS; ++j) {
      float dA = __expf(dtv * A[j]);
      h[j] = dA * h[j] + dtx * Bs[t * kDS + j];
      yv += h[j] * Cs[t * kDS + j];
    }
    yv += xv * Dc;
    float zv = z[((size_t)(b * kL + t)) * (2 * kDI) + c];   // z = xz + 2048
    yv *= zv * sigmoidf_(zv);
    y[idx] = yv; yb[idx] = f2bf(yv);
  }
}

// ---------------------------------------------------------------------------
// Mean over L then head projection: out[b] = mean_t(tok[b,t,:]) . head_w + head_b
// ---------------------------------------------------------------------------
__global__ __launch_bounds__(256) void fm_head(const float* __restrict__ tok,
                                               const float* __restrict__ hw,
                                               const float* __restrict__ hb,
                                               float* __restrict__ out) {
  __shared__ float sb[256];
  const int b = blockIdx.x, tid = threadIdx.x;
  float acc = 0.f;
  for (int d = tid; d < kDM; d += 256) {
    float s = 0.f;
    for (int t = 0; t < kL; ++t) s += tok[((size_t)(b * kL + t)) * kDM + d];
    acc += s * hw[d];
  }
  sb[tid] = acc; __syncthreads();
  for (int o = 128; o > 0; o >>= 1) { if (tid < o) sb[tid] += sb[tid + o]; __syncthreads(); }
  if (tid == 0) out[b] = sb[0] * (1.f / (float)kL) + hb[0];
}

// ---------------------------------------------------------------------------
// Orchestration
// ---------------------------------------------------------------------------
extern "C" void kernel_launch(void* const* d_in, const int* in_sizes, int n_in,
                              void* d_out, int out_size, void* d_ws, size_t ws_size,
                              hipStream_t stream) {
  (void)in_sizes; (void)n_in; (void)out_size; (void)ws_size;
  const float* input    = (const float*)d_in[0];
  const float* p_ln1_g  = (const float*)d_in[1];
  const float* p_ln1_b  = (const float*)d_in[2];
  const float* p_w      = (const float*)d_in[3];
  const float* p_b      = (const float*)d_in[4];
  const float* p_ln2_g  = (const float*)d_in[5];
  const float* p_ln2_b  = (const float*)d_in[6];
  const float* f_ln1_g  = (const float*)d_in[7];
  const float* f_ln1_b  = (const float*)d_in[8];
  const float* f_w      = (const float*)d_in[9];
  const float* f_b      = (const float*)d_in[10];
  const float* f_ln2_g  = (const float*)d_in[11];
  const float* f_ln2_b  = (const float*)d_in[12];
  const float* in_proj_w = (const float*)d_in[13];
  const float* conv_w   = (const float*)d_in[14];
  const float* conv_b   = (const float*)d_in[15];
  const float* x_proj_w = (const float*)d_in[16];
  const float* dt_proj_w = (const float*)d_in[17];
  const float* dt_proj_b = (const float*)d_in[18];
  const float* A_log    = (const float*)d_in[19];
  const float* Dp       = (const float*)d_in[20];
  const float* out_proj_w = (const float*)d_in[21];
  const float* head_w   = (const float*)d_in[22];
  const float* head_b   = (const float*)d_in[23];
  float* out = (float*)d_out;

  char* ws = (char*)d_ws;
  size_t off = 0;
  auto take = [&](size_t bytes) -> char* {
    char* p = ws + off;
    off = (off + bytes + 255) & ~(size_t)255;
    return p;
  };
  float*  fraw  = (float*) take((size_t)kBatch * kCtx * 2 * 4);
  __bf16* xn_b  = (__bf16*)take((size_t)kBatch * kNP * kPatch * 2);
  __bf16* fn_b  = (__bf16*)take((size_t)kBatch * kNP * 2 * kPatch * 2);
  float*  xe    = (float*) take((size_t)kBatch * kNP * kDM * 4);
  float*  fe    = (float*) take((size_t)kBatch * kNP * kDM * 4);
  float*  xcat  = (float*) take((size_t)kTok * kDM * 4);
  __bf16* xcatb = (__bf16*)take((size_t)kTok * kDM * 2);
  __bf16* wb_p  = (__bf16*)take((size_t)kPatch * kDM * 2);
  __bf16* wb_f  = (__bf16*)take((size_t)2 * kPatch * kDM * 2);
  __bf16* wb_in = (__bf16*)take((size_t)kDM * 2 * kDI * 2);
  __bf16* wb_x  = (__bf16*)take((size_t)kDI * 96 * 2);
  __bf16* wb_dt = (__bf16*)take((size_t)kDTR * kDI * 2);
  __bf16* wb_o  = (__bf16*)take((size_t)kDI * kDM * 2);
  float*  xz    = (float*) take((size_t)kTok * 2 * kDI * 4);
  float*  xc    = (float*) take((size_t)kTok * kDI * 4);
  __bf16* xc_b  = (__bf16*)take((size_t)kTok * kDI * 2);
  float*  xdbl  = (float*) take((size_t)kTok * 96 * 4);
  __bf16* dt_b  = (__bf16*)take((size_t)kTok * kDTR * 2);
  float*  dtf   = (float*) take((size_t)kTok * kDI * 4);
  float*  yv    = (float*) take((size_t)kTok * kDI * 4);
  __bf16* y_b   = (__bf16*)take((size_t)kTok * kDI * 2);
  float*  tok   = (float*) take((size_t)kTok * kDM * 4);

  auto cvt = [&](const float* s, __bf16* d, size_t n) {
    fm_to_bf16<<<(unsigned)((n + 255) / 256), 256, 0, stream>>>(s, d, n);
  };

  // weights -> bf16 (deterministic, done every call)
  cvt(p_w,        wb_p,  (size_t)kPatch * kDM);
  cvt(f_w,        wb_f,  (size_t)2 * kPatch * kDM);
  cvt(in_proj_w,  wb_in, (size_t)kDM * 2 * kDI);
  cvt(x_proj_w,   wb_x,  (size_t)kDI * 96);
  cvt(dt_proj_w,  wb_dt, (size_t)kDTR * kDI);
  cvt(out_proj_w, wb_o,  (size_t)kDI * kDM);

  // patch path: ln1 -> GEMM(+bias)
  fm_ln_small<<<kBatch * kNP, kPatch, 0, stream>>>(input, p_ln1_g, p_ln1_b, xn_b, kPatch);
  // freq path: FFT -> ln1 -> GEMM(+bias)
  fm_fft4096<<<kBatch, 256, 0, stream>>>(input, fraw);
  fm_ln_small<<<kBatch * kNP, 2 * kPatch, 0, stream>>>(fraw, f_ln1_g, f_ln1_b, fn_b, 2 * kPatch);
  fm_gemm_bf16<1, false><<<dim3(32, 16), 256, 0, stream>>>(
      xn_b, wb_p, xe, p_b, kBatch * kNP, kDM, kPatch);
  fm_gemm_bf16<1, false><<<dim3(32, 16), 256, 0, stream>>>(
      fn_b, wb_f, fe, f_b, kBatch * kNP, kDM, 2 * kPatch);
  // ln2 + posemb -> concatenated token matrix (f32 + bf16)
  fm_ln_pe<<<kBatch * kNP, 256, 0, stream>>>(xe, p_ln2_g, p_ln2_b, xcat, xcatb, 0);
  fm_ln_pe<<<kBatch * kNP, 256, 0, stream>>>(fe, f_ln2_g, f_ln2_b, xcat, xcatb, 1);

  // mamba
  fm_gemm_bf16<0, false><<<dim3(64, 64), 256, 0, stream>>>(           // in_proj
      xcatb, wb_in, xz, nullptr, kTok, 2 * kDI, kDM);
  fm_conv_silu<<<(kTok * kDI) / 256, 256, 0, stream>>>(xz, conv_w, conv_b, xc, xc_b);
  fm_gemm_bf16<0, true><<<dim3(64, 2), 256, 0, stream>>>(             // x_proj (N=96)
      xc_b, wb_x, xdbl, nullptr, kTok, 96, kDI);
  fm_split_dt<<<(kTok * kDTR) / 256, 256, 0, stream>>>(xdbl, dt_b);
  fm_gemm_bf16<2, false><<<dim3(64, 32), 256, 0, stream>>>(           // dt_proj+softplus
      dt_b, wb_dt, dtf, dt_proj_b, kTok, kDI, kDTR);
  {
    dim3 g(kDI / 256, kBatch);
    fm_scan<<<g, 256, 0, stream>>>(dtf, xc, xdbl, xz + kDI, A_log, Dp, yv, y_b);
  }
  fm_gemm_bf16<0, false><<<dim3(64, 16), 256, 0, stream>>>(           // out_proj
      y_b, wb_o, tok, nullptr, kTok, kDM, kDI);

  // mean over L + head
  fm_head<<<kBatch, 256, 0, stream>>>(tok, head_w, head_b, out);
}
